// VQDecoder_5377299055036
// MI455X (gfx1250) — matrix operand order repaired and checked by reading, not compiled, not run on hardware
//
#include <hip/hip_runtime.h>
#include <math.h>

// Model dims
#define A_SEQ 64
#define DIM   512
#define NH    8
#define NL    6
#define NBAT  512
#define HIDD  2048
#define DD2   256
#define DD8   64
#define ATT_SCALE 0.125f
#define LN_EPS 1e-5f

typedef _Float16 v8h  __attribute__((ext_vector_type(8)));
typedef _Float16 v16h __attribute__((ext_vector_type(16)));
typedef float    v8f  __attribute__((ext_vector_type(8)));

#define BM 64
#define BN 128
#define BK 32
#define LDSP 40   // padded LDS row stride in halves (80B: 16B-aligned, conflict-free)

// TDM (Tensor Data Mover) availability: device pass only; falls back to register staging.
#if defined(__HIP_DEVICE_COMPILE__) && __has_builtin(__builtin_amdgcn_tensor_load_to_lds) && __has_builtin(__builtin_amdgcn_s_wait_tensorcnt)
#define HAVE_TDM 1
#else
#define HAVE_TDM 0
#endif

#if HAVE_TDM
typedef unsigned int tdm_u32x4 __attribute__((ext_vector_type(4)));
typedef int          tdm_i32x8 __attribute__((ext_vector_type(8)));
typedef int          tdm_i32x4 __attribute__((ext_vector_type(4)));

// 2D f16 tile load: tileRows x 32 halves, global row stride strideElems (halves),
// zero-fill rows >= tensorRows. LDS dest padded +4 DWORDs per 16 DWORDs => 40-half rows.
__device__ __forceinline__ void tdm_load_2d(unsigned lds, unsigned long long ga,
                                            long strideElems, int tileRows, int tensorRows) {
  tdm_u32x4 g0;
  g0[0] = 1u;                                   // count=1, user descriptor
  g0[1] = lds;                                  // lds_addr (bytes)
  g0[2] = (unsigned)ga;                         // global_addr low
  g0[3] = (unsigned)(ga >> 32) | (2u << 30);    // global_addr high | type=2 (image)
  tdm_i32x8 g1;
  g1[0] = (1 << 16) | (1 << 20) | (3 << 22) | (3 << 25); // data_size=2B, pad_en, every 16 DW pad 4 DW
  g1[1] = (int)(32u << 16);                     // abar=0 | tensor_dim0 = 32
  g1[2] = (int)(((unsigned)tensorRows & 0xFFFFu) << 16);  // tensor_dim0 hi=0 | tensor_dim1 lo
  g1[3] = (int)(((unsigned)tensorRows >> 16) | (32u << 16)); // tensor_dim1 hi | tile_dim0=32
  g1[4] = tileRows;                             // tile_dim1 | tile_dim2=0
  unsigned long long st = (unsigned long long)strideElems;
  g1[5] = (int)(unsigned)st;                    // tensor_dim0_stride lo32
  g1[6] = (int)(st >> 32);                      // stride hi16 | tensor_dim1_stride lo16=0
  g1[7] = 0;
  tdm_i32x4 gz4 = {0, 0, 0, 0};
  tdm_i32x8 gz8 = {0, 0, 0, 0, 0, 0, 0, 0};
  __builtin_amdgcn_tensor_load_to_lds(g0, g1, gz4, gz4, gz8, 0);
}
#endif

struct GemmP {
  const void* A; long lda, aOut, aIn;
  const void* W; long wN, wK, wOut, wIn;
  void*       C; long ldc, cOut, cIn;
  const float* R;      // residual (f32, same addressing as C), mode 2 only
  const float* bias;   // per-n, optional
  const float* gamma;  // per-n, mode 2 only
  float alpha;
  int N, K, inner, tilesN, mode, wContig;  // mode: 0=none, 1=GELU, 2=residual+gamma
  int aHalf, wHalf, cHalf;                 // operand formats: 1 = _Float16, 0 = float
};

__device__ __forceinline__ float gelu_erf(float x) {
  return 0.5f * x * (1.0f + erff(x * 0.70710678118654752f));
}

// Generic batched NT GEMM: C[m,n] = epilogue(alpha * sum_k A[m,k]*W[n,k] + bias[n])
// A K-contiguous (row stride lda); W element (n,k) at n*wN + k*wK.
// Batch z: zo=z/inner, zi=z%inner; base += zo*Out + zi*In per operand.
// Block: 256 threads = 8 waves (2 M x 4 N), each wave owns a 32x32 tile (2x2 WMMA).
// Double-buffered LDS (TDM-staged for f16 operands), one barrier per K-step.
// M % 64 == 0; K % 32 == 0.
__global__ __launch_bounds__(256) void k_gemm(GemmP p) {
  __shared__ _Float16 As[2][BM][LDSP];
  __shared__ _Float16 Ws[2][BN][LDSP];
  const int tid = threadIdx.x;
  const int z   = blockIdx.y;
  const int zo  = z / p.inner, zi = z % p.inner;
  const long aBase = (long)zo * p.aOut + (long)zi * p.aIn;
  const long wBase = (long)zo * p.wOut + (long)zi * p.wIn;
  const long cBase = (long)zo * p.cOut + (long)zi * p.cIn;
  const int tm = (int)blockIdx.x / p.tilesN;
  const int tn = (int)blockIdx.x % p.tilesN;
  const int m0 = tm * BM, n0 = tn * BN;
  const int wave = tid >> 5, lane = tid & 31;
  const int wm = wave >> 2, wn = wave & 3;   // 2x4 wave grid of 32x32 tiles
  const int lr = lane & 15, lh = lane >> 4;
  const int arow = tid >> 2, acol = (tid & 3) * 8;   // A stage: 64 rows x 4 segs of 8
  const int wrow = tid >> 1, wcol = (tid & 1) * 16;  // W stage: 128 rows x 2 segs of 16

  _Float16 sa[8];    // A staging registers
  _Float16 sw[16];   // W staging registers

#if HAVE_TDM
  const int aTDM = p.aHalf;                 // f16 A: stage with Tensor Data Mover
  const int wTDM = p.wHalf & p.wContig;     // f16 contiguous W: stage with TDM
  const int wTenRows = p.N - n0;            // TDM zero-fills rows beyond tensor
#else
  const int aTDM = 0;
  const int wTDM = 0;
#endif

  auto loadA = [&](int k0) {
    const long idx = aBase + (long)(m0 + arow) * p.lda + (long)(k0 + acol);
    if (p.aHalf) {
      const v8h v = *(const v8h*)((const _Float16*)p.A + idx);
      #pragma unroll
      for (int j = 0; j < 8; ++j) sa[j] = v[j];
    } else {
      const float* ap = (const float*)p.A + idx;
      const float4 a0 = ((const float4*)ap)[0];
      const float4 a1 = ((const float4*)ap)[1];
      sa[0]=(_Float16)a0.x; sa[1]=(_Float16)a0.y; sa[2]=(_Float16)a0.z; sa[3]=(_Float16)a0.w;
      sa[4]=(_Float16)a1.x; sa[5]=(_Float16)a1.y; sa[6]=(_Float16)a1.z; sa[7]=(_Float16)a1.w;
    }
  };
  auto loadW = [&](int k0) {
    const int ng = n0 + wrow;
    if (ng < p.N) {
      const long idx = wBase + (long)ng * p.wN + (long)(k0 + wcol) * p.wK;
      if (p.wHalf) {
        const _Float16* wp = (const _Float16*)p.W + idx;
        if (p.wContig) {
          const v8h w0 = ((const v8h*)wp)[0];
          const v8h w1 = ((const v8h*)wp)[1];
          #pragma unroll
          for (int j = 0; j < 8; ++j) { sw[j] = w0[j]; sw[8 + j] = w1[j]; }
        } else {
          #pragma unroll
          for (int j = 0; j < 16; ++j) sw[j] = wp[(long)j * p.wK];
        }
      } else {
        const float* wp = (const float*)p.W + idx;
        if (p.wContig) {
          const float4* w4 = (const float4*)wp;
          const float4 t0 = w4[0], t1 = w4[1], t2 = w4[2], t3 = w4[3];
          sw[0]=(_Float16)t0.x; sw[1]=(_Float16)t0.y; sw[2]=(_Float16)t0.z; sw[3]=(_Float16)t0.w;
          sw[4]=(_Float16)t1.x; sw[5]=(_Float16)t1.y; sw[6]=(_Float16)t1.z; sw[7]=(_Float16)t1.w;
          sw[8]=(_Float16)t2.x; sw[9]=(_Float16)t2.y; sw[10]=(_Float16)t2.z; sw[11]=(_Float16)t2.w;
          sw[12]=(_Float16)t3.x; sw[13]=(_Float16)t3.y; sw[14]=(_Float16)t3.z; sw[15]=(_Float16)t3.w;
        } else {
          #pragma unroll
          for (int j = 0; j < 16; ++j) sw[j] = (_Float16)wp[(long)j * p.wK];
        }
      }
    } else {
      #pragma unroll
      for (int j = 0; j < 16; ++j) sw[j] = (_Float16)0.0f;
    }
  };
  auto storeA = [&](int buf) {
    v8h t;
    #pragma unroll
    for (int j = 0; j < 8; ++j) t[j] = sa[j];
    *(v8h*)&As[buf][arow][acol] = t;
  };
  auto storeW = [&](int buf) {
    v8h t0, t1;
    #pragma unroll
    for (int j = 0; j < 8; ++j) { t0[j] = sw[j]; t1[j] = sw[8 + j]; }
    *(v8h*)&Ws[buf][wrow][wcol] = t0;
    *(v8h*)&Ws[buf][wrow][wcol + 8] = t1;
  };
#if HAVE_TDM
  auto tdmA = [&](int buf, int k0) {
    const unsigned long long ga = (unsigned long long)(const void*)
        ((const _Float16*)p.A + aBase + (long)m0 * p.lda + k0);
    const unsigned lds = (unsigned)(unsigned long long)(void*)&As[buf][0][0];
    tdm_load_2d(lds, ga, p.lda, BM, BM);
  };
  auto tdmW = [&](int buf, int k0) {
    const unsigned long long ga = (unsigned long long)(const void*)
        ((const _Float16*)p.W + wBase + (long)n0 * p.wN + k0);
    const unsigned lds = (unsigned)(unsigned long long)(void*)&Ws[buf][0][0];
    tdm_load_2d(lds, ga, p.wN, BN, wTenRows);
  };
#endif
  // Fragment gather per ISA 16-bit operand striping:
  // A: lane-half lh -> K chunks [lh*8, lh*8+8) and [lh*8+16, lh*8+24)
  auto fragA = [&](int buf, int row) -> v16h {
    v8h lo = *(const v8h*)&As[buf][row][lh * 8];
    v8h hi = *(const v8h*)&As[buf][row][lh * 8 + 16];
    return __builtin_shufflevector(lo, hi, 0,1,2,3,4,5,6,7,8,9,10,11,12,13,14,15);
  };
  // B: lane-half lh -> K chunks [lh*8, lh*8+8) and [lh*8+16, lh*8+24)
  auto fragW = [&](int buf, int row) -> v16h {
    v8h lo = *(const v8h*)&Ws[buf][row][lh * 8];
    v8h hi = *(const v8h*)&Ws[buf][row][lh * 8 + 16];
    return __builtin_shufflevector(lo, hi, 0,1,2,3,4,5,6,7,8,9,10,11,12,13,14,15);
  };

  v8f acc00 = {}, acc01 = {}, acc10 = {}, acc11 = {};

  // Prologue: stage K-step 0 into buffer 0
#if HAVE_TDM
  if (aTDM) { if (wave == 0) tdmA(0, 0); } else { loadA(0); storeA(0); }
  if (wTDM) { if (wave == 1) tdmW(0, 0); } else { loadW(0); storeW(0); }
  if (wave < 2) __builtin_amdgcn_s_wait_tensorcnt(0);
#else
  loadA(0); loadW(0);
  storeA(0); storeW(0);
#endif
  __syncthreads();

  const int steps = p.K / BK;
  for (int ks = 0; ks < steps; ++ks) {
    const int cur = ks & 1;
    const bool more = (ks + 1 < steps);
    if (more) {  // next tile: HBM (or TDM) in flight over the WMMAs below
      const int k1 = (ks + 1) * BK;
#if HAVE_TDM
      if (aTDM) { if (wave == 0) tdmA(cur ^ 1, k1); } else loadA(k1);
      if (wTDM) { if (wave == 1) tdmW(cur ^ 1, k1); } else loadW(k1);
#else
      loadA(k1); loadW(k1);
#endif
    }
    const v16h a0 = fragA(cur, wm * 32 + lr);
    const v16h a1 = fragA(cur, wm * 32 + 16 + lr);
    const v16h b0 = fragW(cur, wn * 32 + lr);
    const v16h b1 = fragW(cur, wn * 32 + 16 + lr);
    acc00 = __builtin_amdgcn_wmma_f32_16x16x32_f16(false, a0, false, b0, (short)0, acc00, false, false);
    acc01 = __builtin_amdgcn_wmma_f32_16x16x32_f16(false, a0, false, b1, (short)0, acc01, false, false);
    acc10 = __builtin_amdgcn_wmma_f32_16x16x32_f16(false, a1, false, b0, (short)0, acc10, false, false);
    acc11 = __builtin_amdgcn_wmma_f32_16x16x32_f16(false, a1, false, b1, (short)0, acc11, false, false);
    if (more) {
      const int nxt = cur ^ 1;
      if (!aTDM) storeA(nxt);
      if (!wTDM) storeW(nxt);
#if HAVE_TDM
      if (wave < 2) __builtin_amdgcn_s_wait_tensorcnt(0);
#endif
    }
    __syncthreads();
  }

  // Epilogue: C/D layout -> element (m = r + 8*lh, n = lr) in VGPR r
  auto epi = [&](float v, int gm, int gn) {
    if (gn < p.N) {
      float x = p.alpha * v;
      if (p.bias)  x += p.bias[gn];
      if (p.mode == 1) x = gelu_erf(x);
      const long ci = cBase + (long)gm * p.ldc + gn;
      if (p.cHalf) {
        ((_Float16*)p.C)[ci] = (_Float16)x;
      } else {
        if (p.mode == 2) x = p.R[ci] + p.gamma[gn] * x;
        ((float*)p.C)[ci] = x;
      }
    }
  };
  const int gmb = m0 + wm * 32;
  const int gn0 = n0 + wn * 32 + lr;
  const int gn1 = gn0 + 16;
  #pragma unroll
  for (int r = 0; r < 8; ++r) {
    const int gm0 = gmb + r + lh * 8;
    const int gm1 = gm0 + 16;
    epi(acc00[r], gm0, gn0);
    epi(acc01[r], gm0, gn1);
    epi(acc10[r], gm1, gn0);
    epi(acc11[r], gm1, gn1);
  }
}

// LayerNorm over last dim (512): one 64-thread block per row. f32 in, f16 out.
__global__ __launch_bounds__(64) void k_layernorm(const float* __restrict__ x,
                                                  const float* __restrict__ w,
                                                  const float* __restrict__ b,
                                                  _Float16* __restrict__ y) {
  const long row = blockIdx.x;
  const float* xr = x + row * DIM;
  _Float16* yr = y + row * DIM;
  __shared__ float red[64];
  const int t = threadIdx.x;
  float v[8];
  float s = 0.f;
  #pragma unroll
  for (int j = 0; j < 8; ++j) { v[j] = xr[t + j * 64]; s += v[j]; }
  red[t] = s; __syncthreads();
  for (int st = 32; st > 0; st >>= 1) { if (t < st) red[t] += red[t + st]; __syncthreads(); }
  const float mean = red[0] * (1.f / (float)DIM);
  __syncthreads();
  float sq = 0.f;
  #pragma unroll
  for (int j = 0; j < 8; ++j) { const float d = v[j] - mean; sq += d * d; }
  red[t] = sq; __syncthreads();
  for (int st = 32; st > 0; st >>= 1) { if (t < st) red[t] += red[t + st]; __syncthreads(); }
  const float rstd = rsqrtf(red[0] * (1.f / (float)DIM) + LN_EPS);
  #pragma unroll
  for (int j = 0; j < 8; ++j) {
    const int d = t + j * 64;
    yr[d] = (_Float16)((v[j] - mean) * rstd * w[d] + b[d]);
  }
}

// Talking heads: pre-mix (pl), softmax over t (f32 math), post-mix (pw). f16 in/out.
__global__ __launch_bounds__(64) void k_talking(const _Float16* __restrict__ sc_in,
                                                const float* __restrict__ plw,
                                                const float* __restrict__ plb,
                                                const float* __restrict__ pww,
                                                const float* __restrict__ pwb,
                                                _Float16* __restrict__ out) {
  const int bs = blockIdx.x;
  const int b = bs >> 6, s = bs & 63;
  const int t = threadIdx.x;
  __shared__ float sc[NH][64];
  __shared__ float pr[NH][64];
  __shared__ float red[64];
  #pragma unroll
  for (int h = 0; h < NH; ++h)
    sc[h][t] = (float)sc_in[(((long)(b * NH + h)) * 64 + s) * 64 + t];
  __syncthreads();
  for (int g = 0; g < NH; ++g) {
    float v = plb[g];
    #pragma unroll
    for (int h = 0; h < NH; ++h) v += plw[g * NH + h] * sc[h][t];
    red[t] = v; __syncthreads();
    for (int st = 32; st > 0; st >>= 1) {
      if (t < st) red[t] = fmaxf(red[t], red[t + st]);
      __syncthreads();
    }
    const float mx = red[0]; __syncthreads();
    const float e = expf(v - mx);
    red[t] = e; __syncthreads();
    for (int st = 32; st > 0; st >>= 1) {
      if (t < st) red[t] += red[t + st];
      __syncthreads();
    }
    const float sm = red[0]; __syncthreads();
    pr[g][t] = e / sm;
  }
  __syncthreads();
  #pragma unroll
  for (int g2 = 0; g2 < NH; ++g2) {
    float v = pwb[g2];
    #pragma unroll
    for (int g = 0; g < NH; ++g) v += pww[g2 * NH + g] * pr[g][t];
    out[(((long)(b * NH + g2)) * 64 + s) * 64 + t] = (_Float16)v;
  }
}

// GRU gate update for step s: gi (f16), gh (f32), state h (f32); writes h and hs (f32).
__global__ __launch_bounds__(256) void k_gru_gate(const _Float16* __restrict__ gi,
                                                  const float* __restrict__ gh,
                                                  float* __restrict__ h,
                                                  float* __restrict__ hs, int s) {
  const long i = (long)blockIdx.x * 256 + threadIdx.x;   // over B*D
  const int b = (int)(i >> 9);
  const int d = (int)(i & 511);
  const _Float16* gib = gi + ((long)b * A_SEQ + s) * (3 * DIM);
  const float* ghb = gh + (long)b * (3 * DIM);
  const float ir = (float)gib[d], iz = (float)gib[DIM + d], in_ = (float)gib[2 * DIM + d];
  const float hr = ghb[d], hz = ghb[DIM + d], hn = ghb[2 * DIM + d];
  const float r = 1.f / (1.f + expf(-(ir + hr)));
  const float z = 1.f / (1.f + expf(-(iz + hz)));
  const float n = tanhf(in_ + r * hn);
  const float hp = h[i];
  const float hnew = (1.f - z) * n + z * hp;
  h[i] = hnew;
  hs[((long)b * A_SEQ + s) * DIM + d] = hnew;
}

__global__ __launch_bounds__(256) void k_fill(float* __restrict__ p, float v) {
  p[(long)blockIdx.x * 256 + threadIdx.x] = v;
}

// x = x0 + gamma[d] * hs   (all f32)
__global__ __launch_bounds__(256) void k_axpy(const float* __restrict__ x0,
                                              const float* __restrict__ gamma,
                                              const float* __restrict__ hs,
                                              float* __restrict__ out) {
  const long i = (long)blockIdx.x * 256 + threadIdx.x;
  const int d = (int)(i & 511);
  out[i] = x0[i] + gamma[d] * hs[i];
}

// bonds = it * bw[ga] + bb[ga]   (f16 in/out, ga = index % 256)
__global__ __launch_bounds__(256) void k_bonds_affine(const _Float16* __restrict__ it,
                                                      const float* __restrict__ bw,
                                                      const float* __restrict__ bb,
                                                      _Float16* __restrict__ out) {
  const long i = (long)blockIdx.x * 256 + threadIdx.x;
  const int ga = (int)(i & 255);
  out[i] = (_Float16)((float)it[i] * bw[ga] + bb[ga]);
}

static void launch_gemm(hipStream_t st,
                        const void* A, long lda, long aOut, long aIn, int aHalf,
                        const void* W, long wN, long wK, long wOut, long wIn, int wHalf,
                        void* C, long ldc, long cOut, long cIn, int cHalf,
                        const float* R, const float* bias, const float* gamma,
                        float alpha, int M, int N, int K, int inner, int batch, int mode) {
  GemmP p;
  p.A = A; p.lda = lda; p.aOut = aOut; p.aIn = aIn;
  p.W = W; p.wN = wN; p.wK = wK; p.wOut = wOut; p.wIn = wIn;
  p.C = C; p.ldc = ldc; p.cOut = cOut; p.cIn = cIn;
  p.R = R; p.bias = bias; p.gamma = gamma; p.alpha = alpha;
  p.N = N; p.K = K; p.inner = inner; p.tilesN = (N + BN - 1) / BN; p.mode = mode;
  p.wContig = (wK == 1) ? 1 : 0;
  p.aHalf = aHalf; p.wHalf = wHalf; p.cHalf = cHalf;
  dim3 grid((unsigned)((M / BM) * p.tilesN), (unsigned)batch, 1);
  hipLaunchKernelGGL(k_gemm, grid, dim3(256, 1, 1), 0, st, p);
}

extern "C" void kernel_launch(void* const* d_in, const int* in_sizes, int n_in,
                              void* d_out, int out_size, void* d_ws, size_t ws_size,
                              hipStream_t stream) {
  (void)in_sizes; (void)n_in; (void)out_size; (void)ws_size;

  // ---- inputs (JAX pytree order: dicts sorted alphabetically, lists in order) ----
  const float* atoms = (const float*)d_in[0];
  auto blk = [&](int l, int j) -> const float* { return (const float*)d_in[1 + l * 17 + j]; };
  // per-block order: fc1_b fc1_w fc2_b fc2_w g1 g2 ln1_b ln1_w ln2_b ln2_w
  //                  pl_b pl_w proj_b proj_w pw_b pw_w qkv_w
  const float* gamma_rnn = (const float*)d_in[103];
  const float* gru_b_hh  = (const float*)d_in[104];
  const float* gru_b_ih  = (const float*)d_in[105];
  const float* gru_w_hh  = (const float*)d_in[106];
  const float* gru_w_ih  = (const float*)d_in[107];
  const float* rc_atoms_b = (const float*)d_in[108];
  const float* rc_atoms_w = (const float*)d_in[109];
  const float* rc_bb      = (const float*)d_in[110];
  const float* rc_bw      = (const float*)d_in[111];
  const float* rc_deemb_b = (const float*)d_in[112];
  const float* rc_deemb_w = (const float*)d_in[113];
  const float* rc_fact_b  = (const float*)d_in[114];
  const float* rc_fact_w  = (const float*)d_in[115];
  const float* rc_med_b   = (const float*)d_in[116];
  const float* rc_med_w   = (const float*)d_in[117];

  // ---- workspace layout (bytes) ----
  char* wsb = (char*)d_ws;
  float*    bx   = (float*)wsb;                              // [B,S,D] f32        64 MB
  _Float16* ba   = (_Float16*)(wsb + 67108864);              // big f16 buffer    128 MB
  _Float16* bb2  = (_Float16*)(wsb + 67108864 + 134217728);  // [B,S,D] f16        32 MB
  char*     bscB = wsb + 67108864 + 134217728 + 33554432;    // hs f32 / scores f16  64 MB
  float*    hsB  = (float*)bscB;
  _Float16* bsc  = (_Float16*)bscB;
  _Float16* bs2  = (_Float16*)(bscB + 67108864);             // mixed attn f16     32 MB
  float*    bh   = (float*)(bscB + 67108864 + 33554432);     // GRU h [B,D] f32     1 MB
  float*    bgh  = (float*)(bscB + 67108864 + 33554432 + 1048576); // GRU gh f32    3 MB
  // recon overlays inside ba (free after the last fc2); offsets in f16 elements
  _Float16* bav  = ba;                     // [B,S,256]
  _Float16* bbv  = ba + (long)8388608;     // [B,S,64]
  _Float16* bit  = ba + (long)10485760;    // [B,S,4,64]
  _Float16* bbo  = ba + (long)18874368;    // [B,S,4,64]
  float* out_atoms = (float*)d_out;                  // [B,S,15]   491,520
  float* out_bonds = out_atoms + (long)491520;       // [B,4,S,S] 8,388,608

  const int M = NBAT * A_SEQ;  // 32768

  // ---- GRU: gi = x0 @ w_ih^T + b_ih  -> ba f16 [B,S,3D] ----
  launch_gemm(stream, atoms, DIM, 0, 0, 0, gru_w_ih, DIM, 1, 0, 0, 0,
              ba, 3 * DIM, 0, 0, 1, nullptr, gru_b_ih, nullptr,
              1.f, M, 3 * DIM, DIM, 1, 1, 0);
  hipLaunchKernelGGL(k_fill, dim3(1024), dim3(256), 0, stream, bh, 0.f);
  for (int s = 0; s < A_SEQ; ++s) {
    launch_gemm(stream, bh, DIM, 0, 0, 0, gru_w_hh, DIM, 1, 0, 0, 0,
                bgh, 3 * DIM, 0, 0, 0, nullptr, gru_b_hh, nullptr,
                1.f, NBAT, 3 * DIM, DIM, 1, 1, 0);
    hipLaunchKernelGGL(k_gru_gate, dim3(1024), dim3(256), 0, stream, ba, bgh, bh, hsB, s);
  }
  // x = x0 + gamma_rnn * hs
  hipLaunchKernelGGL(k_axpy, dim3(65536), dim3(256), 0, stream, atoms, gamma_rnn, hsB, bx);

  // ---- transformer blocks ----
  for (int l = 0; l < NL; ++l) {
    const float* fc1_b = blk(l, 0),  *fc1_w = blk(l, 1);
    const float* fc2_b = blk(l, 2),  *fc2_w = blk(l, 3);
    const float* g1    = blk(l, 4),  *g2    = blk(l, 5);
    const float* ln1_b = blk(l, 6),  *ln1_w = blk(l, 7);
    const float* ln2_b = blk(l, 8),  *ln2_w = blk(l, 9);
    const float* pl_b  = blk(l, 10), *pl_w  = blk(l, 11);
    const float* proj_b= blk(l, 12), *proj_w= blk(l, 13);
    const float* pw_b  = blk(l, 14), *pw_w  = blk(l, 15);
    const float* qkv_w = blk(l, 16);

    hipLaunchKernelGGL(k_layernorm, dim3(M), dim3(64), 0, stream, bx, ln1_w, ln1_b, bb2);
    // qkv: f16 [B,S,1536]
    launch_gemm(stream, bb2, DIM, 0, 0, 1, qkv_w, DIM, 1, 0, 0, 0,
                ba, 3 * DIM, 0, 0, 1, nullptr, nullptr, nullptr,
                1.f, M, 3 * DIM, DIM, 1, 1, 0);
    // scores[b,h,s,t] = SCALE * q . k   (batch = B*H, inner = H; f16 A/W/C)
    launch_gemm(stream, ba, 1536, 98304, 64, 1,
                ba + 512, 1536, 1, 98304, 64, 1,
                bsc, 64, 32768, 4096, 1, nullptr, nullptr, nullptr,
                ATT_SCALE, 64, 64, 64, NH, NBAT * NH, 0);
    hipLaunchKernelGGL(k_talking, dim3(M), dim3(64), 0, stream, bsc, pl_w, pl_b, pw_w, pw_b, bs2);
    // out[b,s,h*64+d] = sum_t attn2[b,h,s,t] * v[b,t,h,d]   (W = V^T, strided)
    launch_gemm(stream, bs2, 64, 32768, 4096, 1,
                ba + 1024, 1, 1536, 98304, 64, 1,
                bb2, DIM, 32768, 64, 1, nullptr, nullptr, nullptr,
                1.f, 64, 64, 64, NH, NBAT * NH, 0);
    // x = x + g1 * (out @ proj_w^T + proj_b)
    launch_gemm(stream, bb2, DIM, 0, 0, 1, proj_w, DIM, 1, 0, 0, 0,
                bx, DIM, 0, 0, 0, bx, proj_b, g1,
                1.f, M, DIM, DIM, 1, 1, 2);
    hipLaunchKernelGGL(k_layernorm, dim3(M), dim3(64), 0, stream, bx, ln2_w, ln2_b, bb2);
    // fc1 + GELU -> f16
    launch_gemm(stream, bb2, DIM, 0, 0, 1, fc1_w, DIM, 1, 0, 0, 0,
                ba, HIDD, 0, 0, 1, nullptr, fc1_b, nullptr,
                1.f, M, HIDD, DIM, 1, 1, 1);
    // x = x + g2 * (h @ fc2_w^T + fc2_b)
    launch_gemm(stream, ba, HIDD, 0, 0, 1, fc2_w, HIDD, 1, 0, 0, 0,
                bx, DIM, 0, 0, 0, bx, fc2_b, g2,
                1.f, M, DIM, HIDD, 1, 1, 2);
  }

  // ---- graph reconstruction ----
  // av = gelu(x @ deemb^T + b) -> f16
  launch_gemm(stream, bx, DIM, 0, 0, 0, rc_deemb_w, DIM, 1, 0, 0, 0,
              bav, DD2, 0, 0, 1, nullptr, rc_deemb_b, nullptr,
              1.f, M, DD2, DIM, 1, 1, 1);
  // p_atoms = av @ atoms_w^T + b  (N=15, edge-guarded; f32 out)
  launch_gemm(stream, bav, DD2, 0, 0, 1, rc_atoms_w, DD2, 1, 0, 0, 0,
              out_atoms, 15, 0, 0, 0, nullptr, rc_atoms_b, nullptr,
              1.f, M, 15, DD2, 1, 1, 0);
  // bv = x @ fact^T + b -> f16
  launch_gemm(stream, bx, DIM, 0, 0, 0, rc_fact_w, DIM, 1, 0, 0, 0,
              bbv, DD8, 0, 0, 1, nullptr, rc_fact_b, nullptr,
              1.f, M, DD8, DIM, 1, 1, 0);
  // it[b,s,g,a] = bv @ med_w[g,a,:]^T + med_b -> f16
  launch_gemm(stream, bbv, DD8, 0, 0, 1, rc_med_w, DD8, 1, 0, 0, 0,
              bit, 256, 0, 0, 1, nullptr, rc_med_b, nullptr,
              1.f, M, 256, DD8, 1, 1, 0);
  // bonds = it * bw + bb (f16)
  hipLaunchKernelGGL(k_bonds_affine, dim3(32768), dim3(256), 0, stream, bit, rc_bw, rc_bb, bbo);
  // p_bonds[b,g,s,t] = sum_a bonds[b,s,g,a] * it[b,t,g,a]   (batch = B*4, inner = 4; f32 out)
  launch_gemm(stream, bbo, 256, 16384, 64, 1,
              bit, 256, 1, 16384, 64, 1,
              out_bonds, 64, 16384, 4096, 0, nullptr, nullptr, nullptr,
              1.f, 64, 64, 64, 4, NBAT * 4, 0);
}
